// ExchangeBlock_61942018343187
// MI455X (gfx1250) — matrix operand order repaired and checked by reading, not compiled, not run on hardware
//
#include <hip/hip_runtime.h>
#include <hip/hip_bf16.h>

// CDNA5 (gfx1250) fused edge-MLP kernel: gathers + RBF + LayerNorm in VALU,
// all three MLP heads via v_wmma_f32_16x16x32_f16 (16 edges = one M tile).

typedef _Float16 v8h  __attribute__((ext_vector_type(8)));
typedef _Float16 v16h __attribute__((ext_vector_type(16)));
typedef float    v8f  __attribute__((ext_vector_type(8)));

#define WAVES_PER_BLOCK 4
#define THREADS (WAVES_PER_BLOCK * 32)

union H16 { v16h v; v8h h[2]; };

struct alignas(16) SMem {
    float    lng[64];                         // layernorm gain (padded)
    float    lnb[64];                         // layernorm bias
    _Float16 w1t[(32 + 64 + 32) * 64];        // [n][k] K padded 60->64: wn1t | we1t | wr1t
    _Float16 w2t[16 * 128];                   // [n][k] block-diagonal 2nd layer
    _Float16 htile[WAVES_PER_BLOCK][16 * 64]; // per-wave layernormed features (row-major)
    _Float16 sall[WAVES_PER_BLOCK][16 * 128]; // per-wave silu(layer1) concat (row-major)
    float    dbuf[WAVES_PER_BLOCK][16 * 16];  // per-wave layer2 output tile
};

__device__ __forceinline__ v16h lds_a_frag(const _Float16* p0, const _Float16* p1) {
    // A 16x32 f16 frag: lane holds 8 contiguous K at p0 and 8 contiguous K at p1
    H16 u;
    u.h[0] = *(const v8h*)p0;
    u.h[1] = *(const v8h*)p1;
    return u.v;
}

__device__ __forceinline__ v16h lds_b_frag(const _Float16* wt, int rowStride,
                                           int n, int ks, int kh) {
    // B 32x16 f16 frag from N-major (transposed) weights: lane = column n,
    // 8 contiguous K at ks*32 + kh*8 and 8 contiguous K at ks*32 + 16 + kh*8
    const _Float16* p = wt + n * rowStride + ks * 32 + kh * 8;
    H16 u;
    u.h[0] = *(const v8h*)p;
    u.h[1] = *(const v8h*)(p + 16);
    return u.v;
}

__device__ __forceinline__ v8f wmma16(v16h a, v16h b, v8f c) {
    return __builtin_amdgcn_wmma_f32_16x16x32_f16(false, a, false, b,
                                                  (short)0, c, false, false);
}

__device__ __forceinline__ float silu(float x) {
    return x * (1.0f / (1.0f + __expf(-x)));
}

__global__ __launch_bounds__(THREADS)
void fused_edge_mlp_kernel(
    const int* __restrict__ edge_src, const int* __restrict__ edge_dst,
    const int* __restrict__ batch_idx, const float* __restrict__ cell,
    const float* __restrict__ edge_shift, const float* __restrict__ pos,
    const float* __restrict__ angles, const float* __restrict__ cr_i_bonds,
    const float* __restrict__ ln_g, const float* __restrict__ ln_b,
    const float* __restrict__ wn1, const float* __restrict__ bn1,
    const float* __restrict__ wn2, const float* __restrict__ bn2,
    const float* __restrict__ we1, const float* __restrict__ be1,
    const float* __restrict__ we2, const float* __restrict__ be2,
    const float* __restrict__ wr1, const float* __restrict__ br1,
    const float* __restrict__ wr2, const float* __restrict__ br2,
    float* __restrict__ out, int E)
{
    __shared__ SMem sm;
    const int tid  = threadIdx.x;
    const int wave = tid >> 5;
    const int L    = tid & 31;
    const int m    = L & 15;   // M row within tile
    const int kh   = L >> 4;   // K-half selector (A/B frag lane split)
    const int nlo  = L & 15;   // N column within a B/C/D tile

    // ---- Stage weights into LDS (once per block), transposed + K-padded ----
    for (int i = tid; i < 32 * 64; i += THREADS) {        // wn1t
        int n = i >> 6, k = i & 63;
        sm.w1t[i] = (k < 60) ? (_Float16)wn1[k * 32 + n] : (_Float16)0.0f;
    }
    for (int i = tid; i < 64 * 64; i += THREADS) {        // we1t
        int n = i >> 6, k = i & 63;
        sm.w1t[32 * 64 + i] = (k < 60) ? (_Float16)we1[k * 64 + n] : (_Float16)0.0f;
    }
    for (int i = tid; i < 32 * 64; i += THREADS) {        // wr1t
        int n = i >> 6, k = i & 63;
        sm.w1t[96 * 64 + i] = (k < 60) ? (_Float16)wr1[k * 32 + n] : (_Float16)0.0f;
    }
    for (int i = tid; i < 16 * 128; i += THREADS) {       // block-diagonal W2^T
        int n = i >> 7, k = i & 127;
        float v = 0.0f;
        if (n == 0 && k < 32)             v = wn2[k];
        else if (n == 1 && k >= 32 && k < 96) v = we2[k - 32];
        else if (n == 2 && k >= 96)       v = wr2[(k - 96) * 2 + 0];
        else if (n == 3 && k >= 96)       v = wr2[(k - 96) * 2 + 1];
        sm.w2t[i] = (_Float16)v;
    }
    for (int i = tid; i < 60; i += THREADS) {
        sm.lng[i] = ln_g[i];
        sm.lnb[i] = ln_b[i];
    }
    __syncthreads();

    // ---- Per-edge scalar phase: gathers, dist, RBF, LayerNorm -> f16 LDS ----
    const long tileBase = ((long)blockIdx.x * WAVES_PER_BLOCK + wave) * 16;
    const long e  = tileBase + m;
    const size_t ec = (size_t)(e < (long)E ? e : (long)(E - 1));

    const int src = edge_src[ec];
    const int dst = edge_dst[ec];
    const int g   = batch_idx[src];
    const float* C = cell + (size_t)g * 9;
    const float s0 = edge_shift[ec * 3 + 0];
    const float s1 = edge_shift[ec * 3 + 1];
    const float s2 = edge_shift[ec * 3 + 2];
    const float t0 = s0 * C[0] + s1 * C[3] + s2 * C[6];
    const float t1 = s0 * C[1] + s1 * C[4] + s2 * C[7];
    const float t2 = s0 * C[2] + s1 * C[5] + s2 * C[8];
    const float r0 = pos[(size_t)dst * 3 + 0] - pos[(size_t)src * 3 + 0] + t0;
    const float r1 = pos[(size_t)dst * 3 + 1] - pos[(size_t)src * 3 + 1] + t1;
    const float r2 = pos[(size_t)dst * 3 + 2] - pos[(size_t)src * 3 + 2] + t2;
    const float dist = sqrtf(r0 * r0 + r1 * r1 + r2 * r2);

    const float dist_z = (dist - 4.0f) * (1.0f / 0.16f);
    const float ang_z  = (angles[ec] + 0.025f) * (1.0f / 0.06f);
    const float lz0 = (cr_i_bonds[ec * 4 + 0] - 2.8f) * (1.0f / 0.035f);
    const float lz1 = (cr_i_bonds[ec * 4 + 1] - 2.8f) * (1.0f / 0.035f);
    const float lz2 = (cr_i_bonds[ec * 4 + 2] - 2.8f) * (1.0f / 0.035f);
    const float lz3 = (cr_i_bonds[ec * 4 + 3] - 2.8f) * (1.0f / 0.035f);

    // half 0 -> features {dist, angle, leg0}; half 1 -> {leg1, leg2, leg3}
    float fv[3];
    fv[0] = kh ? lz1 : dist_z;
    fv[1] = kh ? lz2 : ang_z;
    fv[2] = kh ? lz3 : lz0;

    float rb[3][10];
    float sum = 0.0f, sq = 0.0f;
#pragma unroll
    for (int f = 0; f < 3; ++f) {
#pragma unroll
        for (int j = 0; j < 10; ++j) {
            float d = fv[f] - (-3.0f + 0.6666666667f * (float)j);
            float r = __expf(-1.125f * d * d);   // COEFF = -0.5/(6/9)^2
            rb[f][j] = r;
            sum += r;
            sq  += r * r;
        }
    }
    sum += __shfl_xor(sum, 16, 32);   // combine the two K-halves of the row
    sq  += __shfl_xor(sq, 16, 32);
    const float mean = sum * (1.0f / 60.0f);
    const float var  = sq * (1.0f / 60.0f) - mean * mean;
    const float rstd = rsqrtf(var + 1e-5f);

    _Float16* hrow = &sm.htile[wave][m * 64];
#pragma unroll
    for (int f = 0; f < 3; ++f) {
#pragma unroll
        for (int j = 0; j < 10; ++j) {
            int k = kh * 30 + f * 10 + j;
            float hv = (rb[f][j] - mean) * rstd * sm.lng[k] + sm.lnb[k];
            hrow[k] = (_Float16)hv;
        }
    }
    if (kh) {  // zero-pad K = 60..63
        hrow[60] = (_Float16)0.0f; hrow[61] = (_Float16)0.0f;
        hrow[62] = (_Float16)0.0f; hrow[63] = (_Float16)0.0f;
    }
    __syncthreads();

    // ---- Layer 1: h[16x64] @ W1[64xN] via WMMA ----
    const int akb = kh ? 8 : 0;   // lane<16: K{0..7,16..23}; lane>=16: K{8..15,24..31}
    const _Float16* hr = &sm.htile[wave][m * 64];
    const v16h a0 = lds_a_frag(hr + akb,      hr + akb + 16);
    const v16h a1 = lds_a_frag(hr + 32 + akb, hr + 32 + akb + 16);
    _Float16* sall = sm.sall[wave];

    // normal head: 60->32 (2 N tiles) -> sall cols [0,32)
#pragma unroll
    for (int t = 0; t < 2; ++t) {
        v8f c = {};
        c = wmma16(a0, lds_b_frag(sm.w1t, 64, t * 16 + nlo, 0, kh), c);
        c = wmma16(a1, lds_b_frag(sm.w1t, 64, t * 16 + nlo, 1, kh), c);
        const int col  = t * 16 + nlo;
        const float bb = bn1[col];
#pragma unroll
        for (int r = 0; r < 8; ++r) {
            int mm = r + kh * 8;
            sall[mm * 128 + col] = (_Float16)silu(c[r] + bb);
        }
    }
    // extreme head: 60->64 (4 N tiles) -> sall cols [32,96)
#pragma unroll
    for (int t = 0; t < 4; ++t) {
        v8f c = {};
        c = wmma16(a0, lds_b_frag(sm.w1t + 32 * 64, 64, t * 16 + nlo, 0, kh), c);
        c = wmma16(a1, lds_b_frag(sm.w1t + 32 * 64, 64, t * 16 + nlo, 1, kh), c);
        const int col  = t * 16 + nlo;
        const float bb = be1[col];
#pragma unroll
        for (int r = 0; r < 8; ++r) {
            int mm = r + kh * 8;
            sall[mm * 128 + 32 + col] = (_Float16)silu(c[r] + bb);
        }
    }
    // routing head: 60->32 (2 N tiles) -> sall cols [96,128)
#pragma unroll
    for (int t = 0; t < 2; ++t) {
        v8f c = {};
        c = wmma16(a0, lds_b_frag(sm.w1t + 96 * 64, 64, t * 16 + nlo, 0, kh), c);
        c = wmma16(a1, lds_b_frag(sm.w1t + 96 * 64, 64, t * 16 + nlo, 1, kh), c);
        const int col  = t * 16 + nlo;
        const float bb = br1[col];
#pragma unroll
        for (int r = 0; r < 8; ++r) {
            int mm = r + kh * 8;
            sall[mm * 128 + 96 + col] = (_Float16)silu(c[r] + bb);
        }
    }
    __syncthreads();

    // ---- Layer 2: s_all[16x128] @ W2[128x16] (block-diag heads) ----
    const _Float16* sr = &sm.sall[wave][m * 128];
    v8f d = {};
#pragma unroll
    for (int ks = 0; ks < 4; ++ks) {
        v16h a = lds_a_frag(sr + ks * 32 + akb, sr + ks * 32 + akb + 16);
        v16h b = lds_b_frag(sm.w2t, 128, nlo, ks, kh);
        d = wmma16(a, b, d);
    }
#pragma unroll
    for (int r = 0; r < 8; ++r)
        sm.dbuf[wave][(r + kh * 8) * 16 + nlo] = d[r];
    __syncthreads();

    // ---- Softmax routing + blend; lanes 0..15 finalize one edge each ----
    if (L < 16) {
        const float* dr = &sm.dbuf[wave][L * 16];
        const float pn = dr[0] + bn2[0];
        const float pe = dr[1] + be2[0];
        const float q0 = dr[2] + br2[0];
        const float q1 = dr[3] + br2[1];
        const float mx = fmaxf(q0, q1);
        const float e0 = __expf(q0 - mx);
        const float e1 = __expf(q1 - mx);
        const float o  = (e0 * pn + e1 * pe) / (e0 + e1);
        if (e < (long)E) out[e] = o;
    }
}

extern "C" void kernel_launch(void* const* d_in, const int* in_sizes, int n_in,
                              void* d_out, int out_size, void* d_ws, size_t ws_size,
                              hipStream_t stream) {
    const int*   edge_src   = (const int*)d_in[0];
    const int*   edge_dst   = (const int*)d_in[1];
    const int*   batch_idx  = (const int*)d_in[2];
    const float* cell       = (const float*)d_in[3];
    const float* edge_shift = (const float*)d_in[4];
    const float* pos        = (const float*)d_in[5];
    const float* angles     = (const float*)d_in[6];
    const float* cr_i_bonds = (const float*)d_in[7];
    const float* ln_g       = (const float*)d_in[8];
    const float* ln_b       = (const float*)d_in[9];
    const float* wn1        = (const float*)d_in[10];
    const float* bn1        = (const float*)d_in[11];
    const float* wn2        = (const float*)d_in[12];
    const float* bn2        = (const float*)d_in[13];
    const float* we1        = (const float*)d_in[14];
    const float* be1        = (const float*)d_in[15];
    const float* we2        = (const float*)d_in[16];
    const float* be2        = (const float*)d_in[17];
    const float* wr1        = (const float*)d_in[18];
    const float* br1        = (const float*)d_in[19];
    const float* wr2        = (const float*)d_in[20];
    const float* br2        = (const float*)d_in[21];

    const int E = in_sizes[0];
    const int edgesPerBlock = WAVES_PER_BLOCK * 16;
    const int blocks = (E + edgesPerBlock - 1) / edgesPerBlock;

    fused_edge_mlp_kernel<<<blocks, THREADS, 0, stream>>>(
        edge_src, edge_dst, batch_idx, cell, edge_shift, pos, angles, cr_i_bonds,
        ln_g, ln_b, wn1, bn1, wn2, bn2, we1, be1, we2, be2, wr1, br1, wr2, br2,
        (float*)d_out, E);
}
